// DPB_Attention_6502580486629
// MI455X (gfx1250) — matrix operand order repaired and checked by reading, not compiled, not run on hardware
//
#include <hip/hip_runtime.h>

typedef __bf16 bf16;
typedef __attribute__((ext_vector_type(16))) __bf16 v16bf;
typedef __attribute__((ext_vector_type(8)))  __bf16 v8bf;
typedef __attribute__((ext_vector_type(4)))  __bf16 v4bf;
typedef __attribute__((ext_vector_type(8)))  float  v8f;
typedef __attribute__((ext_vector_type(4)))  unsigned int v4u;
typedef __attribute__((ext_vector_type(8)))  int v8i;
typedef __attribute__((ext_vector_type(4)))  int v4i;

#define B_TOT   3136
#define NW      49
#define NTOK    64
#define DIM     256
#define HEADS   8
#define HD      32
#define QKV_N   768
#define LBIAS   225
#define POSD    16
#define QSCALE  0.17677669529663687f   // 32^-0.5

// ---- WMMA fragment helpers (bf16 16x16x32, wave32) ----------------------
// A (16x32): lane m = row, hi half -> K {0..15} / {16..31}
__device__ __forceinline__ v16bf load_a_frag(const bf16* row, int hi) {
  v8bf lo = *(const v8bf*)(row + 16 * hi);
  v8bf up = *(const v8bf*)(row + 16 * hi + 8);
  v16bf a;
#pragma unroll
  for (int e = 0; e < 8; ++e) { a[e] = lo[e]; a[e + 8] = up[e]; }
  return a;
}
// B (32x16): lane n = col; rowT = B^T row n (contiguous in k); k = e + 16*hi
__device__ __forceinline__ v16bf load_b_frag(const bf16* rowT, int hi) {
  return *(const v16bf*)(rowT + 16 * hi);
}

// ---- Tensor Data Mover: 1-D global -> LDS copy (dword units) ------------
// D# group0: count=1 | lds_addr | global_addr[56:0] | type=2 ("image")
// D# group1: data_size=4B, tensor_dim0 = tile_dim0 = dwords, stride0 = dwords
__device__ __forceinline__ void tdm_copy_to_lds(void* lds, const void* gsrc,
                                                unsigned dwords) {
  unsigned long long ga = (unsigned long long)gsrc;
  v4u g0;
  g0[0] = 1u;                                        // count=1, user mode
  g0[1] = (unsigned)(unsigned long long)lds;         // LDS byte offset
  g0[2] = (unsigned)ga;
  g0[3] = (unsigned)((ga >> 32) & 0x01FFFFFFull) | (2u << 30);
  v8i g1;
  g1[0] = (int)(2u << 16);                           // data_size = 4 bytes
  g1[1] = (int)((dwords & 0xFFFFu) << 16);           // tensor_dim0[15:0]
  g1[2] = (int)(((dwords >> 16) & 0xFFFFu) | (1u << 16)); // td0 hi | tensor_dim1=1
  g1[3] = (int)((dwords & 0xFFFFu) << 16);           // tile_dim0 = dwords
  g1[4] = 1;                                         // tile_dim1 = 1
  g1[5] = (int)dwords;                               // tensor_dim0_stride lo
  g1[6] = 0;
  g1[7] = 0;
  v4i z4 = {};
#if defined(__clang_major__) && __clang_major__ >= 23
  v8i z8 = {};
  __builtin_amdgcn_tensor_load_to_lds(g0, g1, z4, z4, z8, 0);
#else
  __builtin_amdgcn_tensor_load_to_lds(g0, g1, z4, z4, 0);
#endif
}

// ======================= 1) DynamicPosBias MLP ===========================
__global__ __launch_bounds__(256)
void dpb_kernel(const float* __restrict__ biases,
                const float* __restrict__ ppw, const float* __restrict__ ppb,
                const float* __restrict__ g1, const float* __restrict__ b1,
                const float* __restrict__ w1, const float* __restrict__ fb1,
                const float* __restrict__ g2, const float* __restrict__ b2,
                const float* __restrict__ w2, const float* __restrict__ fb2,
                const float* __restrict__ g3, const float* __restrict__ b3,
                const float* __restrict__ w3, const float* __restrict__ fb3,
                float* __restrict__ ptab) {
  int i = threadIdx.x;
  if (i >= LBIAS) return;
  float bx = biases[i * 2 + 0], by = biases[i * 2 + 1];
  float p[POSD], t[POSD];
#pragma unroll
  for (int j = 0; j < POSD; ++j)
    p[j] = ppw[j * 2] * bx + ppw[j * 2 + 1] * by + ppb[j];

  auto ln_relu = [&](const float* g, const float* b) {
    float m = 0.f;
#pragma unroll
    for (int j = 0; j < POSD; ++j) m += p[j];
    m *= (1.f / POSD);
    float v = 0.f;
#pragma unroll
    for (int j = 0; j < POSD; ++j) { float d = p[j] - m; v += d * d; }
    v *= (1.f / POSD);
    float inv = rsqrtf(v + 1e-5f);
#pragma unroll
    for (int j = 0; j < POSD; ++j) {
      float u = (p[j] - m) * inv * g[j] + b[j];
      t[j] = u > 0.f ? u : 0.f;
    }
  };

  ln_relu(g1, b1);
#pragma unroll
  for (int o = 0; o < POSD; ++o) {
    float s = fb1[o];
#pragma unroll
    for (int j = 0; j < POSD; ++j) s += w1[o * POSD + j] * t[j];
    p[o] = s;
  }
  ln_relu(g2, b2);
#pragma unroll
  for (int o = 0; o < POSD; ++o) {
    float s = fb2[o];
#pragma unroll
    for (int j = 0; j < POSD; ++j) s += w2[o * POSD + j] * t[j];
    p[o] = s;
  }
  ln_relu(g3, b3);
#pragma unroll
  for (int h = 0; h < HEADS; ++h) {
    float s = fb3[h];
#pragma unroll
    for (int j = 0; j < POSD; ++j) s += w3[h * POSD + j] * t[j];
    ptab[i * HEADS + h] = s;
  }
}

// ======================= 2) QKV GEMM (bf16 WMMA) =========================
// grid (3136, 12): block = 64 tokens x 64 output cols, K = 256
__global__ __launch_bounds__(256)
void qkv_gemm_kernel(const float* __restrict__ x,
                     const float* __restrict__ qkv_w,
                     const float* __restrict__ qkv_b,
                     bf16* __restrict__ qkv_out) {
  __shared__ __attribute__((aligned(32))) bf16 Xs[NTOK * DIM];
  __shared__ __attribute__((aligned(32))) bf16 Ws[64 * DIM];
  int b = blockIdx.x, nt = blockIdx.y, tid = threadIdx.x;

  const float* xsrc = x + (size_t)b * NTOK * DIM;
  const float* wsrc = qkv_w + (size_t)nt * 64 * DIM;
  for (int i = tid; i < NTOK * DIM / 4; i += 256) {
    float4 vx = ((const float4*)xsrc)[i];
    float4 vw = ((const float4*)wsrc)[i];
    v4bf px = { (bf16)vx.x, (bf16)vx.y, (bf16)vx.z, (bf16)vx.w };
    v4bf pw = { (bf16)vw.x, (bf16)vw.y, (bf16)vw.z, (bf16)vw.w };
    *(v4bf*)(Xs + i * 4) = px;
    *(v4bf*)(Ws + i * 4) = pw;
  }
  __syncthreads();

  int wave = tid >> 5, lane = tid & 31, hi = lane >> 4, ln = lane & 15;
#pragma unroll
  for (int s = 0; s < 2; ++s) {
    int st = wave * 2 + s, mi = st >> 2, ni = st & 3;
    v8f acc = {};
    const bf16* arow = Xs + (mi * 16 + ln) * DIM;
    const bf16* brow = Ws + (ni * 16 + ln) * DIM;
#pragma unroll
    for (int kc = 0; kc < DIM / 32; ++kc) {
      v16bf a = load_a_frag(arow + kc * 32, hi);
      v16bf bm = load_b_frag(brow + kc * 32, hi);
      acc = __builtin_amdgcn_wmma_f32_16x16x32_bf16(
          false, a, false, bm, (short)0, acc, false, false);
    }
    int c = nt * 64 + ni * 16 + ln;          // 0..767
    int t = c >> 8, ch = c & 255, h = ch >> 5, d = ch & 31;
    float bias = qkv_b[c];
#pragma unroll
    for (int r = 0; r < 8; ++r) {
      int m = mi * 16 + r + 8 * hi;          // token
      float val = acc[r] + bias;
      if (t == 0) val *= QSCALE;
      size_t idx = ((((size_t)t * B_TOT + b) * HEADS + h) * NTOK + m) * HD + d;
      qkv_out[idx] = (bf16)val;
    }
  }
}

// ======================= 3) Attention (per window,head) ==================
__global__ __launch_bounds__(32)
void attn_kernel(const bf16* __restrict__ qkv,
                 const float* __restrict__ ptab,
                 const int* __restrict__ rel_idx,
                 const float* __restrict__ mask,
                 bf16* __restrict__ attn_out) {
  __shared__ __attribute__((aligned(32))) bf16  Qs[NTOK * HD];
  __shared__ __attribute__((aligned(32))) bf16  Ks[NTOK * HD];
  __shared__ __attribute__((aligned(32))) bf16  VTs[HD * NTOK];   // transposed
  __shared__ __attribute__((aligned(32))) float Ss[NTOK * NTOK];
  __shared__ __attribute__((aligned(32))) bf16  Ps[NTOK * NTOK];
  __shared__ __attribute__((aligned(32))) float Pl[LBIAS * HEADS];

  int bh = blockIdx.x, b = bh >> 3, h = bh & 7, w = b % NW;
  int lane = threadIdx.x, hi = lane >> 4, ln = lane & 15;

  const size_t tstride = (size_t)B_TOT * HEADS * NTOK * HD;
  const bf16* qg = qkv + ((size_t)b * HEADS + h) * NTOK * HD;
  const bf16* kg = qg + tstride;
  const bf16* vg = kg + tstride;

  // TDM async copies: q tile, k tile, pos-bias table -> LDS
  tdm_copy_to_lds(Qs, qg, NTOK * HD / 2);          // 1024 dwords
  tdm_copy_to_lds(Ks, kg, NTOK * HD / 2);          // 1024 dwords
  tdm_copy_to_lds(Pl, ptab, LBIAS * HEADS);        // 1800 dwords

  // v is staged transposed manually (TDM cannot transpose)
  for (int i = lane; i < NTOK * HD; i += 32) {
    int tok = i >> 5, d = i & 31;
    VTs[d * NTOK + tok] = vg[i];
  }
  __builtin_amdgcn_s_wait_tensorcnt(0);
  __syncthreads();

  // S = q * k^T  (16 subtiles, K=32 -> one WMMA each)
#pragma unroll
  for (int mi = 0; mi < 4; ++mi)
#pragma unroll
    for (int ni = 0; ni < 4; ++ni) {
      v8f acc = {};
      v16bf a = load_a_frag(Qs + (mi * 16 + ln) * HD, hi);
      v16bf bm = load_b_frag(Ks + (ni * 16 + ln) * HD, hi);
      acc = __builtin_amdgcn_wmma_f32_16x16x32_bf16(
          false, a, false, bm, (short)0, acc, false, false);
#pragma unroll
      for (int r = 0; r < 8; ++r)
        Ss[(mi * 16 + r + 8 * hi) * NTOK + ni * 16 + ln] = acc[r];
    }
  __syncthreads();

  // softmax rows (2 rows per lane), add pos-bias + mask in fp32
  const float* mrow = mask + (size_t)w * NTOK * NTOK;
#pragma unroll
  for (int rr = 0; rr < 2; ++rr) {
    int n = lane * 2 + rr;
    float mx = -3.0e38f;
    for (int m = 0; m < NTOK; ++m) {
      float val = Ss[n * NTOK + m]
                + Pl[rel_idx[n * NTOK + m] * HEADS + h]
                + mrow[n * NTOK + m];
      Ss[n * NTOK + m] = val;
      mx = fmaxf(mx, val);
    }
    float sum = 0.f;
    for (int m = 0; m < NTOK; ++m) sum += __expf(Ss[n * NTOK + m] - mx);
    float inv = 1.f / sum;
    for (int m = 0; m < NTOK; ++m)
      Ps[n * NTOK + m] = (bf16)(__expf(Ss[n * NTOK + m] - mx) * inv);
  }
  __syncthreads();

  // out = P(64x64) @ v(64x32): 8 subtiles x 2 K-chunks
#pragma unroll
  for (int mi = 0; mi < 4; ++mi)
#pragma unroll
    for (int ni = 0; ni < 2; ++ni) {
      v8f acc = {};
      const bf16* arow = Ps + (mi * 16 + ln) * NTOK;
      const bf16* brow = VTs + (ni * 16 + ln) * NTOK;
#pragma unroll
      for (int kc = 0; kc < 2; ++kc) {
        v16bf a = load_a_frag(arow + kc * 32, hi);
        v16bf bm = load_b_frag(brow + kc * 32, hi);
        acc = __builtin_amdgcn_wmma_f32_16x16x32_bf16(
            false, a, false, bm, (short)0, acc, false, false);
      }
#pragma unroll
      for (int r = 0; r < 8; ++r) {
        int m = mi * 16 + r + 8 * hi;
        attn_out[((size_t)b * NTOK + m) * DIM + h * HD + ni * 16 + ln] = (bf16)acc[r];
      }
    }
}

// ======================= 4) Output projection ============================
// grid (3136, 4): 64 tokens x 64 cols, K = 256, fp32 output + bias
__global__ __launch_bounds__(256)
void proj_gemm_kernel(const bf16* __restrict__ attn_in,
                      const float* __restrict__ proj_w,
                      const float* __restrict__ proj_b,
                      float* __restrict__ out) {
  __shared__ __attribute__((aligned(32))) bf16 As[NTOK * DIM];
  __shared__ __attribute__((aligned(32))) bf16 Ws[64 * DIM];
  int b = blockIdx.x, nt = blockIdx.y, tid = threadIdx.x;
  int wave = tid >> 5, lane = tid & 31, hi = lane >> 4, ln = lane & 15;

  // activation tile is already bf16 in global: one TDM copy from wave 0
  const bf16* asrc = attn_in + (size_t)b * NTOK * DIM;
  if (wave == 0 && lane == 0)
    tdm_copy_to_lds(As, asrc, NTOK * DIM / 2);     // 8192 dwords = 32 KB

  const float* wsrc = proj_w + (size_t)nt * 64 * DIM;
  for (int i = tid; i < 64 * DIM / 4; i += 256) {
    float4 vw = ((const float4*)wsrc)[i];
    v4bf pw = { (bf16)vw.x, (bf16)vw.y, (bf16)vw.z, (bf16)vw.w };
    *(v4bf*)(Ws + i * 4) = pw;
  }
  if (wave == 0) __builtin_amdgcn_s_wait_tensorcnt(0);
  __syncthreads();

#pragma unroll
  for (int s = 0; s < 2; ++s) {
    int st = wave * 2 + s, mi = st >> 2, ni = st & 3;
    v8f acc = {};
    const bf16* arow = As + (mi * 16 + ln) * DIM;
    const bf16* brow = Ws + (ni * 16 + ln) * DIM;
#pragma unroll
    for (int kc = 0; kc < DIM / 32; ++kc) {
      v16bf a = load_a_frag(arow + kc * 32, hi);
      v16bf bm = load_b_frag(brow + kc * 32, hi);
      acc = __builtin_amdgcn_wmma_f32_16x16x32_bf16(
          false, a, false, bm, (short)0, acc, false, false);
    }
    int c = nt * 64 + ni * 16 + ln;
    float bias = proj_b[c];
#pragma unroll
    for (int r = 0; r < 8; ++r) {
      int m = mi * 16 + r + 8 * hi;
      out[((size_t)b * NTOK + m) * DIM + c] = acc[r] + bias;
    }
  }
}

// ======================= launch ==========================================
extern "C" void kernel_launch(void* const* d_in, const int* in_sizes, int n_in,
                              void* d_out, int out_size, void* d_ws, size_t ws_size,
                              hipStream_t stream) {
  const float* x       = (const float*)d_in[0];
  const float* mask    = (const float*)d_in[1];
  const float* qkv_w   = (const float*)d_in[2];
  const float* qkv_b   = (const float*)d_in[3];
  const float* proj_w  = (const float*)d_in[4];
  const float* proj_b  = (const float*)d_in[5];
  const float* ppw     = (const float*)d_in[6];
  const float* ppb     = (const float*)d_in[7];
  const float* ln1_g   = (const float*)d_in[8];
  const float* ln1_b   = (const float*)d_in[9];
  const float* fc1_w   = (const float*)d_in[10];
  const float* fc1_b   = (const float*)d_in[11];
  const float* ln2_g   = (const float*)d_in[12];
  const float* ln2_b   = (const float*)d_in[13];
  const float* fc2_w   = (const float*)d_in[14];
  const float* fc2_b   = (const float*)d_in[15];
  const float* ln3_g   = (const float*)d_in[16];
  const float* ln3_b   = (const float*)d_in[17];
  const float* fc3_w   = (const float*)d_in[18];
  const float* fc3_b   = (const float*)d_in[19];
  const float* biases  = (const float*)d_in[20];
  const int*   rel_idx = (const int*)d_in[21];

  // workspace layout
  bf16* qkv = (bf16*)d_ws;                                 // 3*3136*8*64*32 bf16
  size_t qkv_elems = (size_t)3 * B_TOT * HEADS * NTOK * HD;
  bf16* attn = qkv + qkv_elems;                            // 3136*64*256 bf16
  size_t attn_elems = (size_t)B_TOT * NTOK * DIM;
  float* ptab = (float*)(attn + attn_elems);               // 225*8 f32

  dpb_kernel<<<1, 256, 0, stream>>>(biases, ppw, ppb,
                                    ln1_g, ln1_b, fc1_w, fc1_b,
                                    ln2_g, ln2_b, fc2_w, fc2_b,
                                    ln3_g, ln3_b, fc3_w, fc3_b, ptab);
  qkv_gemm_kernel<<<dim3(B_TOT, QKV_N / 64), 256, 0, stream>>>(x, qkv_w, qkv_b, qkv);
  attn_kernel<<<dim3(B_TOT * HEADS), 32, 0, stream>>>(qkv, ptab, rel_idx, mask, attn);
  proj_gemm_kernel<<<dim3(B_TOT, DIM / 64), 256, 0, stream>>>(attn, proj_w, proj_b,
                                                              (float*)d_out);
}
